// RSNN_2_22643067584689
// MI455X (gfx1250) — matrix operand order repaired and checked by reading, not compiled, not run on hardware
//
#include <hip/hip_runtime.h>

typedef __attribute__((ext_vector_type(16))) _Float16 v16h;
typedef __attribute__((ext_vector_type(8)))  _Float16 v8h;
typedef __attribute__((ext_vector_type(8)))  float    v8f;
typedef __attribute__((ext_vector_type(4)))  int      v4i;

#define T_STEPS 250
#define BATCH   256
#define IN_F    700
#define HID_F   1024
#define OUT_F   20
#define KX_PAD  704                 // IN padded to multiple of 32

#define ALPHA_S 0.9048374180359595f  // exp(-0.001/0.01)
#define ALPHA_N 0.9753099120283326f  // exp(-0.001/0.04)

#define BM 64
#define BN 64
#define LDT 40   // LDS row stride in halves (80 B, 16B-aligned, bank-spread)

#if __has_builtin(__builtin_amdgcn_global_load_async_to_lds_b128) && \
    __has_builtin(__builtin_amdgcn_s_wait_asynccnt)
#define USE_ASYNC_LDS 1
typedef __attribute__((address_space(1))) v4i* gv4i_p;   // global src
typedef __attribute__((address_space(3))) v4i* lv4i_p;   // LDS dst
#else
#define USE_ASYNC_LDS 0
#endif

// ---- vectorized f32 -> f16 LDS staging (16 contiguous elements) ----
__device__ __forceinline__ void cvt16_store(_Float16* __restrict__ dst,
                                            const float* __restrict__ src) {
  const float4* s4 = (const float4*)src;
  float4 f0 = s4[0], f1 = s4[1], f2 = s4[2], f3 = s4[3];
  v8h h0, h1;
  h0[0] = (_Float16)f0.x; h0[1] = (_Float16)f0.y; h0[2] = (_Float16)f0.z; h0[3] = (_Float16)f0.w;
  h0[4] = (_Float16)f1.x; h0[5] = (_Float16)f1.y; h0[6] = (_Float16)f1.z; h0[7] = (_Float16)f1.w;
  h1[0] = (_Float16)f2.x; h1[1] = (_Float16)f2.y; h1[2] = (_Float16)f2.z; h1[3] = (_Float16)f2.w;
  h1[4] = (_Float16)f3.x; h1[5] = (_Float16)f3.y; h1[6] = (_Float16)f3.z; h1[7] = (_Float16)f3.w;
  *(v8h*)dst       = h0;
  *(v8h*)(dst + 8) = h1;
}
__device__ __forceinline__ void cvt8_store(_Float16* __restrict__ dst,
                                           const float* __restrict__ src) {
  const float4* s4 = (const float4*)src;
  float4 f0 = s4[0], f1 = s4[1];
  v8h h0;
  h0[0] = (_Float16)f0.x; h0[1] = (_Float16)f0.y; h0[2] = (_Float16)f0.z; h0[3] = (_Float16)f0.w;
  h0[4] = (_Float16)f1.x; h0[5] = (_Float16)f1.y; h0[6] = (_Float16)f1.z; h0[7] = (_Float16)f1.w;
  *(v8h*)dst = h0;
}

// ---- ISA-layout fragment loaders (CDNA5 wave32) ----
// A 16x32 f16: lane m=lane&15, g=lane>>4; v0..3 hold k=g*8+0..7, v4..7 hold k=16+g*8+0..7
__device__ __forceinline__ v16h load_frag_a(const _Float16* __restrict__ tile, int lane) {
  const int m = lane & 15, g = lane >> 4;
  const _Float16* p = tile + m * LDT + g * 8;
  v8h lo = *(const v8h*)p;
  v8h hi = *(const v8h*)(p + 16);
  v16h r;
#pragma unroll
  for (int e = 0; e < 8; ++e) { r[e] = lo[e]; r[e + 8] = hi[e]; }
  return r;
}
// B 32x16 f16: lane n=lane&15, g=lane>>4; v0..3 hold k=g*8+0..7, v4..7 hold k=16+g*8+0..7
__device__ __forceinline__ v16h load_frag_b(const _Float16* __restrict__ tile, int lane) {
  const int n = lane & 15, g = lane >> 4;
  const _Float16* p = tile + n * LDT + g * 8;
  v8h lo = *(const v8h*)p;
  v8h hi = *(const v8h*)(p + 16);
  v16h r;
#pragma unroll
  for (int e = 0; e < 8; ++e) { r[e] = lo[e]; r[e + 8] = hi[e]; }
  return r;
}

__device__ __forceinline__ void do_mma(const _Float16* __restrict__ As,
                                       const _Float16* __restrict__ Bs,
                                       int lane, int waveM, int waveN,
                                       v8f (&acc)[2][2]) {
  v16h a0 = load_frag_a(&As[(waveM +  0) * LDT], lane);
  v16h a1 = load_frag_a(&As[(waveM + 16) * LDT], lane);
  v16h b0 = load_frag_b(&Bs[(waveN +  0) * LDT], lane);
  v16h b1 = load_frag_b(&Bs[(waveN + 16) * LDT], lane);
  acc[0][0] = __builtin_amdgcn_wmma_f32_16x16x32_f16(false, a0, false, b0, (short)0, acc[0][0], false, false);
  acc[0][1] = __builtin_amdgcn_wmma_f32_16x16x32_f16(false, a0, false, b1, (short)0, acc[0][1], false, false);
  acc[1][0] = __builtin_amdgcn_wmma_f32_16x16x32_f16(false, a1, false, b0, (short)0, acc[1][0], false, false);
  acc[1][1] = __builtin_amdgcn_wmma_f32_16x16x32_f16(false, a1, false, b1, (short)0, acc[1][1], false, false);
}

// ============================================================
// Per-timestep kernel: C[b,h] = x_t @ W_in^T + z_prev @ W_rec^T
// fused with LIF update of syn/mem/z + spike-history write.
// ============================================================
__global__ __launch_bounds__(128)
void rsnn_step(const float* __restrict__ x,      // [T,B,IN]
               const float* __restrict__ w_in,   // [HID,IN]
               const float* __restrict__ w_rec,  // [HID,HID]
               float* __restrict__ spk,          // [B,T,HID] (output region)
               float* __restrict__ syn,          // [B,HID] state
               float* __restrict__ mem,          // [B,HID] state
               _Float16* __restrict__ zst,       // [B,HID] f16 spike state
               int t)
{
  __shared__ _Float16 As[BM * LDT];
  __shared__ _Float16 Bs[BN * LDT];

  const int tid  = threadIdx.x;
  const int lane = tid & 31;
  const int wave = tid >> 5;
  const int blockN = blockIdx.x * BN;   // hidden offset
  const int blockM = blockIdx.y * BM;   // batch offset
  const int waveM = (wave & 1) * 32;
  const int waveN = (wave >> 1) * 32;

  const v8f vzero = {0.f,0.f,0.f,0.f,0.f,0.f,0.f,0.f};
  v8f acc[2][2];
  acc[0][0] = vzero; acc[0][1] = vzero; acc[1][0] = vzero; acc[1][1] = vzero;

  const int lr = tid >> 1;          // 0..63 : row staged by this thread
  const int lc = (tid & 1) * 16;    // 0/16  : col offset

  const float*    xrow  = x     + ((size_t)t * BATCH + blockM + lr) * IN_F;
  const float*    wirow = w_in  + (size_t)(blockN + lr) * IN_F;
  const float*    wrrow = w_rec + (size_t)(blockN + lr) * HID_F;
  const _Float16* zrow  = zst   + (size_t)(blockM + lr) * HID_F;

  // ---- phase 1: input chunks fully inside IN_F (k = 0..671) ----
  for (int k0 = 0; k0 < 672; k0 += 32) {
    cvt16_store(&As[lr * LDT + lc], xrow  + k0 + lc);
    cvt16_store(&Bs[lr * LDT + lc], wirow + k0 + lc);
    __syncthreads();
    do_mma(As, Bs, lane, waveM, waveN, acc);
    __syncthreads();
  }

  // ---- phase 2: boundary chunk k = 672..703 (valid to 699, pad to 703) ----
  {
    if ((tid & 1) == 0) {           // k 672..687 : fully valid
      cvt16_store(&As[lr * LDT], xrow  + 672);
      cvt16_store(&Bs[lr * LDT], wirow + 672);
    } else {                        // k 688..699 valid, 700..703 zero
#pragma unroll
      for (int c = 0; c < 12; ++c) {
        As[lr * LDT + 16 + c] = (_Float16)xrow[688 + c];
        Bs[lr * LDT + 16 + c] = (_Float16)wirow[688 + c];
      }
#pragma unroll
      for (int c = 12; c < 16; ++c) {
        As[lr * LDT + 16 + c] = (_Float16)0.f;
        Bs[lr * LDT + 16 + c] = (_Float16)0.f;
      }
    }
    __syncthreads();
    do_mma(As, Bs, lane, waveM, waveN, acc);
    __syncthreads();
  }

  // ---- phase 3: recurrent chunks (z is already f16 -> pure async copy) ----
  for (int kz = 0; kz < HID_F; kz += 32) {
#if USE_ASYNC_LDS
    {
      gv4i_p g0 = (gv4i_p)(zrow + kz + lc);
      lv4i_p l0 = (lv4i_p)(&As[lr * LDT + lc]);
      __builtin_amdgcn_global_load_async_to_lds_b128(g0,     l0,     0, 0);
      __builtin_amdgcn_global_load_async_to_lds_b128(g0 + 1, l0 + 1, 0, 0);
    }
#else
    {
      const uint4* s = (const uint4*)(zrow + kz + lc);
      *(uint4*)(&As[lr * LDT + lc])     = s[0];
      *(uint4*)(&As[lr * LDT + lc + 8]) = s[1];
    }
#endif
    cvt16_store(&Bs[lr * LDT + lc], wrrow + kz + lc);
#if USE_ASYNC_LDS
    __builtin_amdgcn_s_wait_asynccnt(0);
#endif
    __syncthreads();
    do_mma(As, Bs, lane, waveM, waveN, acc);
    __syncthreads();
  }

  // ---- fused LIF epilogue (C/D layout: m = r + (lane>>4)*8, n = lane&15) ----
  const int g = lane >> 4, nl = lane & 15;
#pragma unroll
  for (int mi = 0; mi < 2; ++mi) {
#pragma unroll
    for (int nj = 0; nj < 2; ++nj) {
#pragma unroll
      for (int r = 0; r < 8; ++r) {
        int bb = blockM + waveM + mi * 16 + g * 8 + r;
        int hh = blockN + waveN + nj * 16 + nl;
        size_t idx = (size_t)bb * HID_F + hh;
        float drive = acc[mi][nj][r];
        float s  = (syn[idx] + drive) * ALPHA_S;        // new syn
        float m  = mem[idx];                            // old mem
        float z  = ((m - 1.0f) > 0.f) ? 1.f : 0.f;      // spike from old mem
        float mn = (m * (1.f - z) + s) * ALPHA_N;       // new mem
        syn[idx] = s;
        mem[idx] = mn;
        zst[idx] = (_Float16)z;
        spk[((size_t)bb * T_STEPS + t) * HID_F + hh] = z;
      }
    }
  }
}

// ============================================================
// Deferred output projection: P[r,o] = spk_flat[r,:] @ W_out^T
// r = b*T + t (spk is [B,T,HID] => flat row-major [B*T, HID]).
// ============================================================
__global__ __launch_bounds__(128)
void rsnn_outproj(const float* __restrict__ spk,    // [B*T, HID]
                  const float* __restrict__ w_out,  // [OUT, HID]
                  float* __restrict__ P)            // [B*T, 32]
{
  __shared__ _Float16 As[64 * LDT];
  __shared__ _Float16 Bs[32 * LDT];

  const int tid  = threadIdx.x;
  const int lane = tid & 31;
  const int wave = tid >> 5;
  const int rowM = blockIdx.x * 64;

  const int lr = tid >> 1, lc = (tid & 1) * 16;
  const int br = tid >> 2, bc = (tid & 3) * 8;
  const bool bvalid = (br < OUT_F);

  // zero the padded weight rows once (o = 20..31)
  if (!bvalid) {
    v8h z8;
#pragma unroll
    for (int e = 0; e < 8; ++e) z8[e] = (_Float16)0.f;
    *(v8h*)(&Bs[br * LDT + bc]) = z8;
  }

  const v8f vzero = {0.f,0.f,0.f,0.f,0.f,0.f,0.f,0.f};
  v8f acc0 = vzero, acc1 = vzero;

  const float* arow = spk + (size_t)(rowM + lr) * HID_F;
  const float* brow = bvalid ? (w_out + (size_t)br * HID_F) : w_out;

  for (int k0 = 0; k0 < HID_F; k0 += 32) {
    cvt16_store(&As[lr * LDT + lc], arow + k0 + lc);
    if (bvalid) cvt8_store(&Bs[br * LDT + bc], brow + k0 + bc);
    __syncthreads();

    v16h a  = load_frag_a(&As[(wave * 16) * LDT], lane);
    v16h b0 = load_frag_b(&Bs[0], lane);
    v16h b1 = load_frag_b(&Bs[16 * LDT], lane);
    acc0 = __builtin_amdgcn_wmma_f32_16x16x32_f16(false, a, false, b0, (short)0, acc0, false, false);
    acc1 = __builtin_amdgcn_wmma_f32_16x16x32_f16(false, a, false, b1, (short)0, acc1, false, false);

    __syncthreads();
  }

  const int g = lane >> 4, nl = lane & 15;
#pragma unroll
  for (int r = 0; r < 8; ++r) {
    int m = rowM + wave * 16 + g * 8 + r;
    P[(size_t)m * 32 + nl]      = acc0[r];
    P[(size_t)m * 32 + 16 + nl] = acc1[r];
  }
}

// ============================================================
// Tiny causal scan: out_t = (out_{t-1} + P_t) * alpha_s
// ============================================================
__global__ void rsnn_outscan(const float* __restrict__ P,   // [B*T, 32]
                             float* __restrict__ out_hist)  // [B,T,OUT]
{
  const int b = blockIdx.x;
  const int o = threadIdx.x;
  if (o >= OUT_F) return;
  float s = 0.f;
  for (int t = 0; t < T_STEPS; ++t) {
    s = (s + P[((size_t)b * T_STEPS + t) * 32 + o]) * ALPHA_S;
    out_hist[((size_t)b * T_STEPS + t) * OUT_F + o] = s;
  }
}

extern "C" void kernel_launch(void* const* d_in, const int* in_sizes, int n_in,
                              void* d_out, int out_size, void* d_ws, size_t ws_size,
                              hipStream_t stream) {
  const float* x     = (const float*)d_in[0];  // [T,B,IN]
  const float* w_in  = (const float*)d_in[1];  // [HID,IN]
  const float* w_rec = (const float*)d_in[2];  // [HID,HID]
  const float* w_out = (const float*)d_in[3];  // [OUT,HID]

  float* out_hist = (float*)d_out;                                   // [B,T,OUT]
  float* spk      = (float*)d_out + (size_t)BATCH * T_STEPS * OUT_F; // [B,T,HID]

  // Workspace layout
  char* ws = (char*)d_ws;
  float*    syn = (float*)(ws);                                      // 1 MB
  float*    mem = (float*)(ws + (size_t)BATCH * HID_F * 4);          // 1 MB
  _Float16* zst = (_Float16*)(ws + (size_t)2 * BATCH * HID_F * 4);   // 0.5 MB
  const size_t state_bytes = (size_t)BATCH * HID_F * (4 + 4 + 2);
  float*    P   = (float*)(ws + state_bytes);                        // [B*T,32] = 8 MB

  // Zero-init recurrence state (syn, mem, z) every call.
  (void)hipMemsetAsync(d_ws, 0, state_bytes, stream);

  // Serial recurrence: one fused GEMM+LIF kernel per timestep.
  dim3 gridStep(HID_F / BN, BATCH / BM);   // (16, 4) = 64 WGs
  for (int t = 0; t < T_STEPS; ++t) {
    rsnn_step<<<gridStep, 128, 0, stream>>>(x, w_in, w_rec, spk, syn, mem, zst, t);
  }

  // Deferred output projection (fully parallel WMMA GEMM) + causal scan.
  rsnn_outproj<<<(BATCH * T_STEPS) / 64, 128, 0, stream>>>(spk, w_out, P);
  rsnn_outscan<<<BATCH, 32, 0, stream>>>(P, out_hist);
}
